// BitNetAttention_3350074491564
// MI455X (gfx1250) — 16-bit and 8-bit matrix operand order repaired and checked by reading, compiled, not run on hardware
//
#include <hip/hip_runtime.h>

typedef _Float16 v16h __attribute__((ext_vector_type(16)));
typedef float    v8f  __attribute__((ext_vector_type(8)));
typedef int      v8i  __attribute__((ext_vector_type(8)));

union Frag8  { v8i  v; int i[8]; unsigned u[8]; };
union FragH  { v16h v; unsigned u[8]; };
union AccF   { v8f  v; float f[8]; };

#define NB    2
#define SEQ   2048
#define DMOD  1024
#define HEADS 16
#define DK    64
#define MROWS (NB*SEQ)
#define NACT  (MROWS*DMOD)
#define NWEL  (DMOD*DMOD)

#if defined(__has_builtin)
#if __has_builtin(__builtin_amdgcn_tensor_load_to_lds) && __has_builtin(__builtin_amdgcn_s_wait_tensorcnt)
#define USE_TDM 1
#endif
#endif

#ifdef USE_TDM
typedef unsigned int u32x4 __attribute__((ext_vector_type(4)));
typedef int          i32x4 __attribute__((ext_vector_type(4)));
typedef int          i32x8 __attribute__((ext_vector_type(8)));
typedef __attribute__((address_space(3))) const char lds_cchar;

__device__ __forceinline__ unsigned __lds_off(const void* p) {
    return (unsigned)(size_t)(lds_cchar*)p;
}

__device__ __forceinline__ void tdm_load_2d(const void* gsrc, const void* ldst,
                                            unsigned tile_w, unsigned tile_h,
                                            unsigned long long stride_b) {
    unsigned long long ga = (unsigned long long)gsrc;
    u32x4 g0;
    g0[0] = 1u;
    g0[1] = __lds_off(ldst);
    g0[2] = (unsigned)(ga & 0xffffffffull);
    g0[3] = (unsigned)((ga >> 32) & 0x01ffffffull)
          | (2u << 30);
    i32x8 g1;
    g1[0] = 0;
    g1[1] = (int)((tile_w & 0xffffu) << 16);
    g1[2] = (int)(((tile_w >> 16) & 0xffffu) | ((tile_h & 0xffffu) << 16));
    g1[3] = (int)(((tile_h >> 16) & 0xffffu) | ((tile_w & 0xffffu) << 16));
    g1[4] = (int)(tile_h & 0xffffu);
    g1[5] = (int)(unsigned)(stride_b & 0xffffffffull);
    g1[6] = (int)(unsigned)((stride_b >> 32) & 0xffffull);
    g1[7] = 0;
    i32x4 gz = {0, 0, 0, 0};
#if defined(__clang_major__) && __clang_major__ >= 23
    i32x8 gz8 = {0, 0, 0, 0, 0, 0, 0, 0};
    __builtin_amdgcn_tensor_load_to_lds(g0, g1, gz, gz, gz8, 0);
#else
    __builtin_amdgcn_tensor_load_to_lds(g0, g1, gz, gz, 0);
#endif
}
#endif

__global__ void zero_sc_kernel(float* sc) {
    if (threadIdx.x < 64) sc[threadIdx.x] = 0.0f;
}

__global__ __launch_bounds__(256) void absmax_kernel(const float* __restrict__ x,
                                                     int n, float* sc, int idx) {
    __shared__ unsigned red[256];
    unsigned mx = 0u;
    for (int i = blockIdx.x * 256 + threadIdx.x; i < n; i += gridDim.x * 256)
        mx = max(mx, __float_as_uint(fabsf(x[i])));
    red[threadIdx.x] = mx; __syncthreads();
    for (int s = 128; s > 0; s >>= 1) {
        if (threadIdx.x < s) red[threadIdx.x] = max(red[threadIdx.x], red[threadIdx.x + s]);
        __syncthreads();
    }
    if (threadIdx.x == 0) atomicMax((unsigned*)&sc[idx], red[0]);
}

__global__ __launch_bounds__(256) void abssum_p1(const float* __restrict__ x,
                                                 int n, float* part) {
    __shared__ float red[256];
    float s = 0.0f;
    for (int i = blockIdx.x * 256 + threadIdx.x; i < n; i += gridDim.x * 256)
        s += fabsf(x[i]);
    red[threadIdx.x] = s; __syncthreads();
    for (int w = 128; w > 0; w >>= 1) {
        if (threadIdx.x < w) red[threadIdx.x] += red[threadIdx.x + w];
        __syncthreads();
    }
    if (threadIdx.x == 0) part[blockIdx.x] = red[0];
}

__global__ void abssum_p2(const float* __restrict__ part, float* sc, int idx) {
    __shared__ float red[256];
    red[threadIdx.x] = part[threadIdx.x]; __syncthreads();
    for (int w = 128; w > 0; w >>= 1) {
        if (threadIdx.x < w) red[threadIdx.x] += red[threadIdx.x + w];
        __syncthreads();
    }
    if (threadIdx.x == 0) sc[idx] = red[0];
}

__global__ __launch_bounds__(256) void quant_act_kernel(const float* __restrict__ x,
                                                        signed char* __restrict__ xq,
                                                        const float* sc, int idx, int n) {
    int i = blockIdx.x * 256 + threadIdx.x;
    if (i >= n) return;
    float s = sc[idx] * (1.0f / 127.0f);
    float inv = (s > 0.0f) ? (1.0f / s) : 0.0f;
    float r = rintf(x[i] * inv);
    r = fminf(127.0f, fmaxf(-127.0f, r));
    xq[i] = (signed char)(int)r;
}

__global__ __launch_bounds__(256) void quant_wt_kernel(const float* __restrict__ w,
                                                       signed char* __restrict__ wq,
                                                       const float* sc, int idx, int n) {
    int i = blockIdx.x * 256 + threadIdx.x;
    if (i >= n) return;
    float s = sc[idx] * (1.0f / (float)NWEL);
    float inv = (s > 0.0f) ? (1.0f / s) : 0.0f;
    float r = rintf(w[i] * inv);
    r = fminf(1.0f, fmaxf(-1.0f, r));
    wq[i] = (signed char)(int)r;
}

__global__ __launch_bounds__(256) void gemm_iu8_kernel(const signed char* __restrict__ Xq,
                                                       const signed char* __restrict__ Wq,
                                                       float* __restrict__ out,
                                                       const float* __restrict__ sc,
                                                       int aidx, int widx, int mode) {
#ifdef USE_TDM
    __shared__ signed char sX[2][64][64];
    __shared__ signed char sW[2][128][64];
#else
    __shared__ signed char sX[1][64][64];
    __shared__ signed char sW[1][128][64];
#endif
    const int t    = threadIdx.x;
    const int lane = t & 31;
    const int wid  = t >> 5;
    const bool lo  = lane < 16;
    const int waveM = wid & 1;
    const int waveN = wid >> 1;
    const int m0 = blockIdx.x * 64;
    const int n0 = blockIdx.y * 128;

    Frag8 c[2][2];
    #pragma unroll
    for (int a = 0; a < 2; ++a)
        #pragma unroll
        for (int b = 0; b < 2; ++b)
            #pragma unroll
            for (int r = 0; r < 8; ++r) c[a][b].i[r] = 0;

#ifdef USE_TDM
    if (wid == 0) {
        tdm_load_2d(Xq + (size_t)m0 * DMOD, &sX[0][0][0], 64, 64, DMOD);
        tdm_load_2d(Wq + (size_t)n0 * DMOD, &sW[0][0][0], 64, 128, DMOD);
        __builtin_amdgcn_s_wait_tensorcnt(0);
    }
    __syncthreads();
#endif

    for (int k0 = 0; k0 < DMOD; k0 += 64) {
#ifdef USE_TDM
        const int cur = (k0 >> 6) & 1;
        if (wid == 0 && (k0 + 64) < DMOD) {
            tdm_load_2d(Xq + (size_t)m0 * DMOD + k0 + 64, &sX[cur ^ 1][0][0], 64, 64, DMOD);
            tdm_load_2d(Wq + (size_t)n0 * DMOD + k0 + 64, &sW[cur ^ 1][0][0], 64, 128, DMOD);
        }
#else
        const int cur = 0;
        #pragma unroll
        for (int j = 0; j < 4; ++j) {
            int w = t + 256 * j, r = w >> 4, cw = w & 15;
            ((unsigned*)sX)[w] = *(const unsigned*)&Xq[(size_t)(m0 + r) * DMOD + k0 + cw * 4];
        }
        #pragma unroll
        for (int j = 0; j < 8; ++j) {
            int w = t + 256 * j, r = w >> 4, cw = w & 15;
            ((unsigned*)sW)[w] = *(const unsigned*)&Wq[(size_t)(n0 + r) * DMOD + k0 + cw * 4];
        }
        __syncthreads();
#endif
        Frag8 af[2], bf[2];
        #pragma unroll
        for (int mi = 0; mi < 2; ++mi) {
            int row = waveM * 32 + mi * 16 + (lane & 15);
            #pragma unroll
            for (int i = 0; i < 8; ++i) {
                int koff = ((i & 1) << 2) + (((i >> 1) & 1) << 4) + (((i >> 2) & 1) << 5) + (lo ? 0 : 8);
                af[mi].u[i] = *(const unsigned*)&sX[cur][row][koff];
            }
        }
        #pragma unroll
        for (int ni = 0; ni < 2; ++ni) {
            int coln = waveN * 32 + ni * 16 + (lane & 15);
            #pragma unroll
            for (int i = 0; i < 8; ++i) {
                int koff = ((i & 1) << 2) + (((i >> 1) & 1) << 4) + (((i >> 2) & 1) << 5) + (lo ? 0 : 8);
                bf[ni].u[i] = *(const unsigned*)&sW[cur][coln][koff];
            }
        }
        #pragma unroll
        for (int mi = 0; mi < 2; ++mi)
            #pragma unroll
            for (int ni = 0; ni < 2; ++ni)
                c[mi][ni].v = __builtin_amdgcn_wmma_i32_16x16x64_iu8(
                    true, af[mi].v, true, bf[ni].v, c[mi][ni].v, false, false);
#ifdef USE_TDM
        if (wid == 0) __builtin_amdgcn_s_wait_tensorcnt(0);
#endif
        __syncthreads();
    }

    const float scale = (sc[aidx] * (1.0f / 127.0f)) * (sc[widx] * (1.0f / (float)NWEL));
    #pragma unroll
    for (int mi = 0; mi < 2; ++mi) {
        int mbase = m0 + waveM * 32 + mi * 16 + (lo ? 0 : 8);
        #pragma unroll
        for (int ni = 0; ni < 2; ++ni) {
            int n = n0 + waveN * 32 + ni * 16 + (lane & 15);
            #pragma unroll
            for (int r = 0; r < 8; ++r) {
                int m = mbase + r;
                float val = (float)c[mi][ni].i[r] * scale;
                if (mode == 0) {
                    int h = n >> 6, d = n & 63, bb = m >> 11, s = m & 2047;
                    out[(((size_t)(bb * HEADS + h)) * SEQ + s) * DK + d] = val;
                } else {
                    out[(size_t)m * DMOD + n] = val;
                }
            }
        }
    }
}

__global__ __launch_bounds__(256) void rope_cvt_kernel(const float* __restrict__ qf,
                                                       const float* __restrict__ kf,
                                                       const float* __restrict__ vf,
                                                       _Float16* __restrict__ qh,
                                                       _Float16* __restrict__ kh,
                                                       _Float16* __restrict__ vh) {
    int i = blockIdx.x * 256 + threadIdx.x;
    int d = i & 63;
    int s = (i >> 6) & (SEQ - 1);
    int j = d & 31;
    float ang = (float)s * __powf(10000.0f, -(float)(2 * j) * (1.0f / 64.0f));
    float c, sn;
    __sincosf(ang, &sn, &c);
    float q = qf[i], k = kf[i], q2, k2;
    if (d < 32) { q2 = q * c - qf[i + 32] * sn; k2 = k * c - kf[i + 32] * sn; }
    else        { q2 = q * c + qf[i - 32] * sn; k2 = k * c + kf[i - 32] * sn; }
    qh[i] = (_Float16)q2;
    kh[i] = (_Float16)k2;
    vh[i] = (_Float16)vf[i];
}

__global__ __launch_bounds__(256) void scores_kernel(const _Float16* __restrict__ qh,
                                                     const _Float16* __restrict__ kh,
                                                     float* __restrict__ attnW) {
    __shared__ _Float16 sQ[128][64];
    __shared__ _Float16 sK[64][64];
    const int bh = blockIdx.z;
    const int m0 = blockIdx.x * 128;
    const int n0 = blockIdx.y * 64;
    const _Float16* qb = qh + (size_t)bh * SEQ * DK;
    const _Float16* kb = kh + (size_t)bh * SEQ * DK;
    const int t = threadIdx.x, lane = t & 31, wid = t >> 5;
    const bool lo = lane < 16;
    const int waveM = wid & 3;
    const int waveN = wid >> 2;

#ifdef USE_TDM
    if (wid == 0) {
        tdm_load_2d(qb + (size_t)m0 * DK, &sQ[0][0], 128 * DK * 2, 1, 128 * DK * 2);
        tdm_load_2d(kb + (size_t)n0 * DK, &sK[0][0], 64 * DK * 2, 1, 64 * DK * 2);
        __builtin_amdgcn_s_wait_tensorcnt(0);
    }
    __syncthreads();
#else
    #pragma unroll
    for (int j = 0; j < 16; ++j) {
        int w = t + 256 * j, r = w >> 5, cw = w & 31;
        ((unsigned*)sQ)[w] = *(const unsigned*)&qb[(size_t)(m0 + r) * DK + cw * 2];
    }
    #pragma unroll
    for (int j = 0; j < 8; ++j) {
        int w = t + 256 * j, r = w >> 5, cw = w & 31;
        ((unsigned*)sK)[w] = *(const unsigned*)&kb[(size_t)(n0 + r) * DK + cw * 2];
    }
    __syncthreads();
#endif

    AccF c[2][2];
    #pragma unroll
    for (int a = 0; a < 2; ++a)
        #pragma unroll
        for (int b = 0; b < 2; ++b)
            #pragma unroll
            for (int r = 0; r < 8; ++r) c[a][b].f[r] = 0.0f;

    #pragma unroll
    for (int ks = 0; ks < 2; ++ks) {
        FragH af[2], bf[2];
        #pragma unroll
        for (int mi = 0; mi < 2; ++mi) {
            int row = waveM * 32 + mi * 16 + (lane & 15);
            #pragma unroll
            for (int i = 0; i < 8; ++i) {
                int kp = ks * 32 + ((i & 3) << 1) + ((i >> 2) << 4) + (lo ? 0 : 8);
                af[mi].u[i] = *(const unsigned*)&sQ[row][kp];
            }
        }
        #pragma unroll
        for (int ni = 0; ni < 2; ++ni) {
            int coln = waveN * 32 + ni * 16 + (lane & 15);
            #pragma unroll
            for (int i = 0; i < 8; ++i) {
                int kp = ks * 32 + ((i & 3) << 1) + ((i >> 2) << 4) + (lo ? 0 : 8);
                bf[ni].u[i] = *(const unsigned*)&sK[coln][kp];
            }
        }
        #pragma unroll
        for (int mi = 0; mi < 2; ++mi)
            #pragma unroll
            for (int ni = 0; ni < 2; ++ni)
                c[mi][ni].v = __builtin_amdgcn_wmma_f32_16x16x32_f16(
                    false, af[mi].v, false, bf[ni].v, (short)0, c[mi][ni].v, false, false);
    }

    float* ob = attnW + (size_t)bh * SEQ * SEQ;
    #pragma unroll
    for (int mi = 0; mi < 2; ++mi) {
        int mbase = m0 + waveM * 32 + mi * 16 + (lo ? 0 : 8);
        #pragma unroll
        for (int ni = 0; ni < 2; ++ni) {
            int n = n0 + waveN * 32 + ni * 16 + (lane & 15);
            #pragma unroll
            for (int r = 0; r < 8; ++r)
                ob[(size_t)(mbase + r) * SEQ + n] = c[mi][ni].f[r] * 0.125f;
        }
    }
}

__global__ __launch_bounds__(256) void softmax_kernel(float* __restrict__ attnW) {
    const int row = blockIdx.x, bh = blockIdx.y, t = threadIdx.x;
    float* p = attnW + ((size_t)bh * SEQ + row) * SEQ;
    __shared__ float red[256];
    float v[8], mx = -3.4e38f;
    #pragma unroll
    for (int j = 0; j < 8; ++j) { v[j] = p[t + 256 * j]; mx = fmaxf(mx, v[j]); }
    red[t] = mx; __syncthreads();
    for (int s = 128; s > 0; s >>= 1) {
        if (t < s) red[t] = fmaxf(red[t], red[t + s]);
        __syncthreads();
    }
    mx = red[0]; __syncthreads();
    float sum = 0.0f;
    #pragma unroll
    for (int j = 0; j < 8; ++j) { v[j] = __expf(v[j] - mx); sum += v[j]; }
    red[t] = sum; __syncthreads();
    for (int s = 128; s > 0; s >>= 1) {
        if (t < s) red[t] += red[t + s];
        __syncthreads();
    }
    float inv = 1.0f / red[0];
    #pragma unroll
    for (int j = 0; j < 8; ++j) p[t + 256 * j] = v[j] * inv;
}

__global__ __launch_bounds__(256) void attnv_kernel(const float* __restrict__ attnW,
                                                    const _Float16* __restrict__ vh,
                                                    float* __restrict__ ao) {
    __shared__ _Float16 sP[64][32];
    __shared__ _Float16 sV[32][64];
    const int bh = blockIdx.y;
    const int m0 = blockIdx.x * 64;
    const float* pb = attnW + (size_t)bh * SEQ * SEQ;
    const _Float16* vb = vh + (size_t)bh * SEQ * DK;
    const int t = threadIdx.x, lane = t & 31, wid = t >> 5;
    const bool lo = lane < 16;
    const int mi4 = wid & 3;
    const int nhf = wid >> 2;

    AccF c[2];
    #pragma unroll
    for (int a = 0; a < 2; ++a)
        #pragma unroll
        for (int r = 0; r < 8; ++r) c[a].f[r] = 0.0f;

    for (int k0 = 0; k0 < SEQ; k0 += 32) {
#ifdef USE_TDM
        if (wid == 0)
            tdm_load_2d(vb + (size_t)k0 * DK, &sV[0][0], 32 * DK * 2, 1, 32 * DK * 2);
#endif
        #pragma unroll
        for (int j = 0; j < 8; ++j) {
            int idx = t + 256 * j, r = idx >> 5, cc = idx & 31;
            sP[r][cc] = (_Float16)pb[(size_t)(m0 + r) * SEQ + k0 + cc];
        }
#ifdef USE_TDM
        if (wid == 0) __builtin_amdgcn_s_wait_tensorcnt(0);
#else
        #pragma unroll
        for (int j = 0; j < 8; ++j) {
            int idx = t + 256 * j, kk = idx >> 6, n = idx & 63;
            sV[kk][n] = vb[(size_t)(k0 + kk) * DK + n];
        }
#endif
        __syncthreads();

        FragH af, bf[2];
        int row = mi4 * 16 + (lane & 15);
        #pragma unroll
        for (int i = 0; i < 8; ++i) {
            int kp = ((i & 3) << 1) + ((i >> 2) << 4) + (lo ? 0 : 8);
            af.u[i] = *(const unsigned*)&sP[row][kp];
        }
        #pragma unroll
        for (int ni = 0; ni < 2; ++ni) {
            int coln = nhf * 32 + ni * 16 + (lane & 15);
            #pragma unroll
            for (int i = 0; i < 8; ++i) {
                int kk = ((i & 3) << 1) + ((i >> 2) << 4) + (lo ? 0 : 8);
                unsigned l16 = *(const unsigned short*)&sV[kk][coln];
                unsigned h16 = *(const unsigned short*)&sV[kk + 1][coln];
                bf[ni].u[i] = l16 | (h16 << 16);
            }
            c[ni].v = __builtin_amdgcn_wmma_f32_16x16x32_f16(
                false, af.v, false, bf[ni].v, (short)0, c[ni].v, false, false);
        }
        __syncthreads();
    }

    const int b = bh >> 4, h = bh & 15;
    #pragma unroll
    for (int ni = 0; ni < 2; ++ni) {
        int n = nhf * 32 + ni * 16 + (lane & 15);
        int mbase = m0 + mi4 * 16 + (lo ? 0 : 8);
        #pragma unroll
        for (int r = 0; r < 8; ++r) {
            int s = mbase + r;
            ao[((size_t)(b * SEQ + s)) * DMOD + h * DK + n] = c[ni].f[r];
        }
    }
}

extern "C" void kernel_launch(void* const* d_in, const int* in_sizes, int n_in,
                              void* d_out, int out_size, void* d_ws, size_t ws_size,
                              hipStream_t stream) {
    const float* hidden = (const float*)d_in[0];
    const float* Wq = (const float*)d_in[1];
    const float* Wk = (const float*)d_in[2];
    const float* Wv = (const float*)d_in[3];
    const float* Wo = (const float*)d_in[4];

    float* out   = (float*)d_out;
    float* attnW = out + (size_t)MROWS * DMOD;

    char* ws = (char*)d_ws;
    float*       sc   = (float*)ws;
    float*       part = (float*)(ws + 256);
    size_t o = 8192;
    signed char* xq  = (signed char*)(ws + o); o += (size_t)NACT;
    signed char* wqq = (signed char*)(ws + o); o += (size_t)NWEL;
    signed char* wkq = (signed char*)(ws + o); o += (size_t)NWEL;
    signed char* wvq = (signed char*)(ws + o); o += (size_t)NWEL;
    signed char* woq = (signed char*)(ws + o); o += (size_t)NWEL;
    float* qf = (float*)(ws + o); o += (size_t)NACT * 4;
    float* kf = (float*)(ws + o); o += (size_t)NACT * 4;
    float* vf = (float*)(ws + o); o += (size_t)NACT * 4;
    _Float16* qh = (_Float16*)(ws + o); o += (size_t)NACT * 2;
    _Float16* kh = (_Float16*)(ws + o); o += (size_t)NACT * 2;
    _Float16* vh = (_Float16*)(ws + o); o += (size_t)NACT * 2;
    float* ao = qf;

    zero_sc_kernel<<<1, 64, 0, stream>>>(sc);
    absmax_kernel<<<1024, 256, 0, stream>>>(hidden, NACT, sc, 0);
    abssum_p1<<<256, 256, 0, stream>>>(Wq, NWEL, part);
    abssum_p2<<<1, 256, 0, stream>>>(part, sc, 1);
    abssum_p1<<<256, 256, 0, stream>>>(Wk, NWEL, part);
    abssum_p2<<<1, 256, 0, stream>>>(part, sc, 2);
    abssum_p1<<<256, 256, 0, stream>>>(Wv, NWEL, part);
    abssum_p2<<<1, 256, 0, stream>>>(part, sc, 3);
    abssum_p1<<<256, 256, 0, stream>>>(Wo, NWEL, part);
    abssum_p2<<<1, 256, 0, stream>>>(part, sc, 4);

    quant_act_kernel<<<NACT / 256, 256, 0, stream>>>(hidden, xq, sc, 0, NACT);
    quant_wt_kernel<<<NWEL / 256, 256, 0, stream>>>(Wq, wqq, sc, 1, NWEL);
    quant_wt_kernel<<<NWEL / 256, 256, 0, stream>>>(Wk, wkq, sc, 2, NWEL);
    quant_wt_kernel<<<NWEL / 256, 256, 0, stream>>>(Wv, wvq, sc, 3, NWEL);
    quant_wt_kernel<<<NWEL / 256, 256, 0, stream>>>(Wo, woq, sc, 4, NWEL);

    dim3 gg(MROWS / 64, DMOD / 128);
    gemm_iu8_kernel<<<gg, 256, 0, stream>>>(xq, wqq, qf, sc, 0, 1, 0);
    gemm_iu8_kernel<<<gg, 256, 0, stream>>>(xq, wkq, kf, sc, 0, 2, 0);
    gemm_iu8_kernel<<<gg, 256, 0, stream>>>(xq, wvq, vf, sc, 0, 3, 0);

    rope_cvt_kernel<<<NACT / 256, 256, 0, stream>>>(qf, kf, vf, qh, kh, vh);

    scores_kernel<<<dim3(SEQ / 128, SEQ / 64, NB * HEADS), 256, 0, stream>>>(qh, kh, attnW);
    softmax_kernel<<<dim3(SEQ, NB * HEADS), 256, 0, stream>>>(attnW);
    attnv_kernel<<<dim3(SEQ / 64, NB * HEADS), 256, 0, stream>>>(attnW, vh, ao);

    absmax_kernel<<<1024, 256, 0, stream>>>(ao, NACT, sc, 5);
    quant_act_kernel<<<NACT / 256, 256, 0, stream>>>(ao, xq, sc, 5, NACT);
    gemm_iu8_kernel<<<gg, 256, 0, stream>>>(xq, woq, out, sc, 5, 4, 1);
}
